// changeAttention_64493228917159
// MI455X (gfx1250) — matrix operand order repaired and checked by reading, not compiled, not run on hardware
//
#include <hip/hip_runtime.h>
#include <hip/hip_bf16.h>
#include <stdint.h>

typedef __bf16 bf16;
typedef __attribute__((ext_vector_type(16))) __bf16 v16bf;
typedef __attribute__((ext_vector_type(8)))  __bf16 v8bf;
typedef __attribute__((ext_vector_type(8)))  float  v8f;

typedef __attribute__((ext_vector_type(4))) unsigned int v4u32;
typedef __attribute__((ext_vector_type(8))) int          v8i32;
typedef __attribute__((ext_vector_type(4))) int          v4i32;

// Problem constants (fixed by the reference)
static constexpr int Bb   = 8;     // batch
static constexpr int Nq   = 1024;  // query tokens
static constexpr int Cc   = 512;   // dim
static constexpr int HDS  = 8;     // heads
static constexpr int Dh   = 64;    // head dim
static constexpr int Rr   = 128;   // low-rank bottleneck
static constexpr int Nkv  = 256;   // reduced tokens (16x16)

#if __has_builtin(__builtin_amdgcn_tensor_load_to_lds) && __has_builtin(__builtin_amdgcn_s_wait_tensorcnt)
#define USE_TDM 1
#else
#define USE_TDM 0
#endif

// ---------------------------------------------------------------------------
// f32 -> bf16 elementwise convert
// ---------------------------------------------------------------------------
__global__ void cvt_bf16_kernel(const float* __restrict__ in, bf16* __restrict__ out, int n) {
  int i = blockIdx.x * blockDim.x + threadIdx.x;
  if (i < n) out[i] = (bf16)in[i];
}

// ---------------------------------------------------------------------------
// Depthwise 2x2 stride-2 conv + bias + LayerNorm over C, output bf16.
// One block per output row (b, nn); 256 threads, 2 channels/thread.
// ---------------------------------------------------------------------------
__global__ __launch_bounds__(256)
void convln_kernel(const float* __restrict__ x, const float* __restrict__ sr_w,
                   const float* __restrict__ sr_b, const float* __restrict__ ln_g,
                   const float* __restrict__ ln_b, bf16* __restrict__ xr) {
  __shared__ float rs[256], rs2[256];
  const int b  = blockIdx.x >> 8;
  const int nn = blockIdx.x & 255;
  const int hp = nn >> 4, wp = nn & 15;
  const int t  = threadIdx.x;

  float vals[2];
  float s = 0.f, s2 = 0.f;
#pragma unroll
  for (int j = 0; j < 2; ++j) {
    const int c = t * 2 + j;
    float acc = sr_b[c];
#pragma unroll
    for (int dy = 0; dy < 2; ++dy)
#pragma unroll
      for (int dx = 0; dx < 2; ++dx) {
        const int row = (2 * hp + dy) * 32 + (2 * wp + dx);
        acc += x[((size_t)b * Nq + row) * Cc + c] * sr_w[c * 4 + dy * 2 + dx];
      }
    vals[j] = acc; s += acc; s2 += acc * acc;
  }
  rs[t] = s; rs2[t] = s2;
  __syncthreads();
  for (int off = 128; off > 0; off >>= 1) {
    if (t < off) { rs[t] += rs[t + off]; rs2[t] += rs2[t + off]; }
    __syncthreads();
  }
  const float mu  = rs[0] * (1.0f / 512.0f);
  const float var = rs2[0] * (1.0f / 512.0f) - mu * mu;
  const float inv = rsqrtf(var + 1e-5f);
#pragma unroll
  for (int j = 0; j < 2; ++j) {
    const int c = t * 2 + j;
    const float o = (vals[j] - mu) * inv * ln_g[c] + ln_b[c];
    xr[((size_t)b * Nkv + nn) * Cc + c] = (bf16)o;
  }
}

// ---------------------------------------------------------------------------
// TDM: stage a (64 rows x 32 bf16) chunk of W into LDS via the Tensor Data
// Mover (descriptor per CDNA5 ISA). 2-D tensor: groups
// 2/3 zero (VADDR2/3 disabled). lds_addr = low 32 bits of the flat shared
// pointer (ISA §10.2).
// This toolchain exposes the 6-arg builtin form:
//   (uint32x4 g0, int32x8 g1, int32x4, int32x4, int32x8, i32 cpol)
// ---------------------------------------------------------------------------
#if USE_TDM
__device__ __forceinline__
void tdm_stage_w(const bf16* gsrc, uint32_t lds_addr, int rem_cols, int rem_rows,
                 int row_stride_elems) {
  const uint64_t ga = (uint64_t)(uintptr_t)gsrc;
  v4u32 g0;
  g0[0] = 1u;                                   // count=1, user descriptor
  g0[1] = lds_addr;                             // LDS byte address
  g0[2] = (uint32_t)(ga & 0xffffffffu);         // global_addr[31:0]
  g0[3] = (uint32_t)((ga >> 32) & 0x01ffffffu)  // global_addr[56:32]
        | (2u << 30);                           // type = 2 ("image")

  const uint32_t td0 = (uint32_t)rem_cols;      // tensor_dim0 (elems)
  const uint32_t td1 = (uint32_t)rem_rows;      // tensor_dim1 (rows)
  const uint64_t s0  = (uint64_t)row_stride_elems;  // tensor_dim0_stride
  v8i32 g1;
  g1[0] = (int)(1u << 16);                      // wg_mask=0, data_size=1 (2B)
  g1[1] = (int)((td0 & 0xffffu) << 16);         // tensor_dim0[15:0] @ bits 63:48
  g1[2] = (int)(((td0 >> 16) & 0xffffu) | ((td1 & 0xffffu) << 16));
  g1[3] = (int)(((td1 >> 16) & 0xffffu) | (32u << 16));  // tile_dim0 = 32
  g1[4] = (int)64u;                             // tile_dim1 = 64, tile_dim2 = 0
  g1[5] = (int)(uint32_t)(s0 & 0xffffffffu);    // stride0[31:0]
  g1[6] = (int)(uint32_t)((s0 >> 32) & 0xffffu);// stride0[47:32], stride1 lo = 0
  g1[7] = 0;
  const v4i32 z4 = {0, 0, 0, 0};
  const v8i32 z8 = {0, 0, 0, 0, 0, 0, 0, 0};
  __builtin_amdgcn_tensor_load_to_lds(g0, g1, z4, z4, z8, 0);
}
#endif

// ---------------------------------------------------------------------------
// WMMA GEMM: out = A(MxK,bf16) @ W(NxK,bf16)^T [+ bias]
// Requires M%128==0, N%64==0, K%32==0 (true for all launches below).
// Block = 256 (8 waves); wave w owns rows [(by*8+w)*16, +16), cols [bx*64,+64).
// W chunks (64x32) are staged into LDS once per block (TDM, double-buffered);
// all 8 waves build B fragments from LDS (ds_load_b128), cutting global B
// traffic 8x. A fragments are direct per-lane global loads + prefetch.
// ---------------------------------------------------------------------------
__global__ __launch_bounds__(256)
void gemm_wmma_kernel(const bf16* __restrict__ A, const bf16* __restrict__ W,
                      float* __restrict__ outF, bf16* __restrict__ outB,
                      const float* __restrict__ bias, int M, int N, int K) {
  __shared__ bf16 wtile[2][64 * 32];   // 2 x 4 KB double buffer

  const int lane = threadIdx.x & 31;
  const int wave = threadIdx.x >> 5;
  const int g  = lane >> 4;
  const int lr = lane & 15;
  const int m0 = (blockIdx.y * 8 + wave) * 16;
  const int n0 = blockIdx.x * 64;

  const bf16* wbase = W + (size_t)n0 * K;
  const bf16* arow  = A + (size_t)(m0 + lr) * K;

#if USE_TDM
  const uint32_t lds_buf[2] = {
    (uint32_t)(uintptr_t)(void*)&wtile[0][0],
    (uint32_t)(uintptr_t)(void*)&wtile[1][0]
  };
  if (wave == 0) tdm_stage_w(wbase, lds_buf[0], K, N - n0, K);
#endif

  v8f acc[4] = {};
  int buf = 0;

  for (int kc = 0; kc < K; kc += 32) {
    const bool more = (kc + 32) < K;

#if USE_TDM
    if (wave == 0) {
      if (more) {
        tdm_stage_w(wbase + (kc + 32), lds_buf[buf ^ 1], K - (kc + 32), N - n0, K);
        __builtin_amdgcn_s_wait_tensorcnt(1);   // current buffer's DMA done
      } else {
        __builtin_amdgcn_s_wait_tensorcnt(0);
      }
    }
    __syncthreads();                            // staged chunk visible to all waves
#else
    __syncthreads();                            // previous reads finished
    {
      // cooperative synchronous staging: 256 threads x 8 bf16 (16B) each
      const int row = threadIdx.x >> 2;
      const int col = (threadIdx.x & 3) * 8;
      *(v8bf*)&wtile[0][row * 32 + col] = *(const v8bf*)(wbase + (size_t)row * K + kc + col);
    }
    __syncthreads();
    buf = 0;
#endif

    if (kc + 64 <= K) __builtin_prefetch(arow + kc + 64, 0, 1);  // global_prefetch_b8
    v8bf alo = *(const v8bf*)(arow + kc + g * 16);
    v8bf ahi = *(const v8bf*)(arow + kc + g * 16 + 8);
    v16bf a;
#pragma unroll
    for (int i = 0; i < 8; ++i) { a[i] = alo[i]; a[i + 8] = ahi[i]; }

    const bf16* wl = &wtile[buf][0];
#pragma unroll
    for (int t = 0; t < 4; ++t) {
      v16bf bb = *(const v16bf*)(wl + (t * 16 + lr) * 32 + g * 16);  // ds_load_b128 x2
      acc[t] = __builtin_amdgcn_wmma_f32_16x16x32_bf16(false, a, false, bb,
                                                       (short)0, acc[t], false, false);
    }

#if USE_TDM
    __syncthreads();                            // all waves done reading buf
    buf ^= 1;
#endif
  }

#pragma unroll
  for (int t = 0; t < 4; ++t) {
    const int n = n0 + t * 16 + lr;
    const float bv = bias ? bias[n] : 0.0f;
#pragma unroll
    for (int r = 0; r < 8; ++r) {
      const int m = m0 + r + g * 8;
      const float vv = acc[t][r] + bv;
      if (outF) outF[(size_t)m * N + n] = vv;
      else      outB[(size_t)m * N + n] = (bf16)vv;
    }
  }
}

// ---------------------------------------------------------------------------
// Fused attention: per block = one (b, head, 16-row q tile).
// Phase 1: 8 waves compute 16x256 scores via WMMA (q·k^T), scale+mask -> LDS.
// Phase 2: 256-thread softmax over 256 cols per row (LDS reductions).
// Phase 3: waves 0..3 compute S·v via WMMA (A-fragments rebuilt from LDS).
// ---------------------------------------------------------------------------
__global__ __launch_bounds__(256)
void attn_wmma_kernel(const bf16* __restrict__ q, const bf16* __restrict__ k,
                      const bf16* __restrict__ v, const uint8_t* __restrict__ mask,
                      bf16* __restrict__ ao) {
  __shared__ float S[16][Nkv + 8];
  __shared__ float red[16][16];
  __shared__ float rowmax[16], rowsum[16];

  const int bid = blockIdx.x;
  const int b   = bid >> 9;
  const int h   = (bid >> 6) & 7;
  const int m0  = (bid & 63) * 16;

  const int tid  = threadIdx.x;
  const int lane = tid & 31;
  const int wave = tid >> 5;
  const int g  = lane >> 4;
  const int lr = lane & 15;

  // ---- Phase 1: S = mask(scale * q k^T) ----
  {
    const bf16* qrow = q + ((size_t)b * Nq + m0 + lr) * Cc + h * Dh;
    v16bf a0, a1;
#pragma unroll
    for (int i = 0; i < 8; ++i) {
      a0[i]     = qrow[g * 16 + i];
      a0[i + 8] = qrow[g * 16 + 8 + i];
      a1[i]     = qrow[32 + g * 16 + i];
      a1[i + 8] = qrow[32 + g * 16 + 8 + i];
    }
#pragma unroll
    for (int tt = 0; tt < 2; ++tt) {
      const int t = wave * 2 + tt;          // n-tile 0..15
      v8f acc = {};
      const bf16* krow = k + ((size_t)b * Nkv + t * 16 + lr) * Cc + h * Dh;
      v16bf b0 = *(const v16bf*)(krow + g * 16);
      v16bf b1 = *(const v16bf*)(krow + 32 + g * 16);
      acc = __builtin_amdgcn_wmma_f32_16x16x32_bf16(false, a0, false, b0, (short)0, acc, false, false);
      acc = __builtin_amdgcn_wmma_f32_16x16x32_bf16(false, a1, false, b1, (short)0, acc, false, false);
#pragma unroll
      for (int r = 0; r < 8; ++r) {
        const int m = r + g * 8;
        const int n = t * 16 + lr;
        float sc = acc[r] * 0.125f;          // 1/sqrt(64)
        if (mask[((size_t)(b * 8 + h) * Nq + m0 + m) * Nkv + n]) sc = -__builtin_inff();
        S[m][n] = sc;
      }
    }
  }
  __syncthreads();

  // ---- Phase 2: softmax over 256 per row (16 threads per row) ----
  {
    const int row = tid >> 4;
    const int seg = tid & 15;
    float lm = -__builtin_inff();
#pragma unroll
    for (int i = 0; i < 16; ++i) lm = fmaxf(lm, S[row][seg * 16 + i]);
    red[row][seg] = lm;
    __syncthreads();
    if (seg == 0) {
      float m = red[row][0];
#pragma unroll
      for (int i = 1; i < 16; ++i) m = fmaxf(m, red[row][i]);
      rowmax[row] = m;
    }
    __syncthreads();
    const float rm = rowmax[row];
    float ls = 0.f;
#pragma unroll
    for (int i = 0; i < 16; ++i) {
      const float e = __expf(S[row][seg * 16 + i] - rm);
      S[row][seg * 16 + i] = e;
      ls += e;
    }
    red[row][seg] = ls;
    __syncthreads();
    if (seg == 0) {
      float s = 0.f;
#pragma unroll
      for (int i = 0; i < 16; ++i) s += red[row][i];
      rowsum[row] = s;
    }
    __syncthreads();
    const float inv = 1.0f / rowsum[row];
#pragma unroll
    for (int i = 0; i < 16; ++i) S[row][seg * 16 + i] *= inv;
  }
  __syncthreads();

  // ---- Phase 3: out = S @ v (waves 0..3, one 16x16 tile each) ----
  if (wave < 4) {
    const int t = wave;                      // dd-tile
    v8f acc = {};
    const bf16* vcol = v + (size_t)b * Nkv * Cc + h * Dh + t * 16 + lr;
    for (int kc = 0; kc < Nkv; kc += 32) {
      v16bf a;
#pragma unroll
      for (int i = 0; i < 8; ++i) {
        a[i]     = (bf16)S[lr][kc + g * 16 + i];
        a[i + 8] = (bf16)S[lr][kc + g * 16 + 8 + i];
      }
      v16bf bb;
#pragma unroll
      for (int i = 0; i < 16; ++i) bb[i] = vcol[(size_t)(kc + g * 16 + i) * Cc];
      acc = __builtin_amdgcn_wmma_f32_16x16x32_bf16(false, a, false, bb, (short)0, acc, false, false);
    }
#pragma unroll
    for (int r = 0; r < 8; ++r) {
      const int m = m0 + r + g * 8;
      ao[((size_t)b * Nq + m) * Cc + h * Dh + t * 16 + lr] = (bf16)acc[r];
    }
  }
}

// ---------------------------------------------------------------------------
// Host-side orchestration
// ---------------------------------------------------------------------------
extern "C" void kernel_launch(void* const* d_in, const int* in_sizes, int n_in,
                              void* d_out, int out_size, void* d_ws, size_t ws_size,
                              hipStream_t stream) {
  const float*   x      = (const float*)d_in[0];
  const uint8_t* mask   = (const uint8_t*)d_in[1];   // jnp bool -> 1 byte
  const float*   q_w1   = (const float*)d_in[2];
  const float*   q_w2   = (const float*)d_in[3];
  const float*   k_w1   = (const float*)d_in[4];
  const float*   k_w2   = (const float*)d_in[5];
  const float*   v_w1   = (const float*)d_in[6];
  const float*   v_w2   = (const float*)d_in[7];
  const float*   sr_w   = (const float*)d_in[8];
  const float*   sr_b   = (const float*)d_in[9];
  const float*   ln_g   = (const float*)d_in[10];
  const float*   ln_b   = (const float*)d_in[11];
  const float*   proj_w = (const float*)d_in[12];
  const float*   proj_b = (const float*)d_in[13];
  float* out = (float*)d_out;

  const int Mq = Bb * Nq;   // 8192
  const int Mk = Bb * Nkv;  // 2048

  // carve workspace (256B aligned chunks)
  char*  ws  = (char*)d_ws;
  size_t off = 0;
  auto carve = [&](size_t bytes) -> char* {
    char* p = ws + off;
    off += (bytes + 255) & ~(size_t)255;
    return p;
  };
  bf16* x_bf   = (bf16*)carve((size_t)Mq * Cc * 2);
  bf16* qw1b   = (bf16*)carve((size_t)Rr * Cc * 2);
  bf16* qw2b   = (bf16*)carve((size_t)Cc * Rr * 2);
  bf16* kw1b   = (bf16*)carve((size_t)Rr * Cc * 2);
  bf16* kw2b   = (bf16*)carve((size_t)Cc * Rr * 2);
  bf16* vw1b   = (bf16*)carve((size_t)Rr * Cc * 2);
  bf16* vw2b   = (bf16*)carve((size_t)Cc * Rr * 2);
  bf16* pwb    = (bf16*)carve((size_t)Cc * Cc * 2);
  bf16* xr_bf  = (bf16*)carve((size_t)Mk * Cc * 2);
  bf16* t_q    = (bf16*)carve((size_t)Mq * Rr * 2);
  bf16* t_k    = (bf16*)carve((size_t)Mk * Rr * 2);
  bf16* t_v    = (bf16*)carve((size_t)Mk * Rr * 2);
  bf16* q_bf   = (bf16*)carve((size_t)Mq * Cc * 2);
  bf16* k_bf   = (bf16*)carve((size_t)Mk * Cc * 2);
  bf16* v_bf   = (bf16*)carve((size_t)Mk * Cc * 2);
  bf16* ao_bf  = (bf16*)carve((size_t)Mq * Cc * 2);

  const dim3 blk(256);

  // 1) bf16 conversions
  cvt_bf16_kernel<<<(Mq * Cc + 255) / 256, blk, 0, stream>>>(x, x_bf, Mq * Cc);
  cvt_bf16_kernel<<<(Rr * Cc + 255) / 256, blk, 0, stream>>>(q_w1, qw1b, Rr * Cc);
  cvt_bf16_kernel<<<(Cc * Rr + 255) / 256, blk, 0, stream>>>(q_w2, qw2b, Cc * Rr);
  cvt_bf16_kernel<<<(Rr * Cc + 255) / 256, blk, 0, stream>>>(k_w1, kw1b, Rr * Cc);
  cvt_bf16_kernel<<<(Cc * Rr + 255) / 256, blk, 0, stream>>>(k_w2, kw2b, Cc * Rr);
  cvt_bf16_kernel<<<(Rr * Cc + 255) / 256, blk, 0, stream>>>(v_w1, vw1b, Rr * Cc);
  cvt_bf16_kernel<<<(Cc * Rr + 255) / 256, blk, 0, stream>>>(v_w2, vw2b, Cc * Rr);
  cvt_bf16_kernel<<<(Cc * Cc + 255) / 256, blk, 0, stream>>>(proj_w, pwb, Cc * Cc);

  // 2) depthwise conv + LayerNorm -> xr (bf16)
  convln_kernel<<<Bb * Nkv, blk, 0, stream>>>(x, sr_w, sr_b, ln_g, ln_b, xr_bf);

  // 3) low-rank projections (all via WMMA + TDM-staged weights)
  gemm_wmma_kernel<<<dim3(Rr / 64, Mq / 128), blk, 0, stream>>>(x_bf, qw1b, nullptr, t_q, nullptr, Mq, Rr, Cc);
  gemm_wmma_kernel<<<dim3(Cc / 64, Mq / 128), blk, 0, stream>>>(t_q, qw2b, nullptr, q_bf, nullptr, Mq, Cc, Rr);
  gemm_wmma_kernel<<<dim3(Rr / 64, Mk / 128), blk, 0, stream>>>(xr_bf, kw1b, nullptr, t_k, nullptr, Mk, Rr, Cc);
  gemm_wmma_kernel<<<dim3(Cc / 64, Mk / 128), blk, 0, stream>>>(t_k, kw2b, nullptr, k_bf, nullptr, Mk, Cc, Rr);
  gemm_wmma_kernel<<<dim3(Rr / 64, Mk / 128), blk, 0, stream>>>(xr_bf, vw1b, nullptr, t_v, nullptr, Mk, Rr, Cc);
  gemm_wmma_kernel<<<dim3(Cc / 64, Mk / 128), blk, 0, stream>>>(t_v, vw2b, nullptr, v_bf, nullptr, Mk, Cc, Rr);

  // 4) fused masked-softmax attention
  attn_wmma_kernel<<<Bb * HDS * (Nq / 16), blk, 0, stream>>>(q_bf, k_bf, v_bf, mask, ao_bf);

  // 5) output projection + bias -> f32 d_out
  gemm_wmma_kernel<<<dim3(Cc / 64, Mq / 128), blk, 0, stream>>>(ao_bf, pwb, out, nullptr, proj_b, Mq, Cc, Cc);
}
